// DySA_10797547782412
// MI455X (gfx1250) — matrix operand order repaired and checked by reading, not compiled, not run on hardware
//
#include <hip/hip_runtime.h>
#include <hip/hip_bf16.h>
#include <math.h>

#define BATCH 2
#define CCH 192
#define NHEAD 6
#define HH 128
#define WW 128
#define HWSZ (HH * WW)
#define NOFF 9

typedef __attribute__((ext_vector_type(16))) __bf16 v16bf;
typedef __attribute__((ext_vector_type(8)))  float  v8f;

__device__ __forceinline__ unsigned int f32_to_bf16_bits(float f) {
  unsigned int u = __float_as_uint(f);
  return (u + 0x7FFFu + ((u >> 16) & 1u)) >> 16;   // round-to-nearest-even
}
__device__ __forceinline__ unsigned int pack_bf16x2(float lo, float hi) {
  return f32_to_bf16_bits(lo) | (f32_to_bf16_bits(hi) << 16);
}

// A-fragment K index (contiguous order, matches B staging):
// lanes 0-15: K 0..15; lanes 16-31: K 16..31.
__device__ __forceinline__ int a_frag_k(int e, int half) {
  return half * 16 + e;
}

// ---------------------------------------------------------------------------
// Weight repack: f32 row-major [Cout x Cin] -> bf16 fragment stream
// [tile][chunk][lane][16e], zero-padded for rows >= Cout. One thread per
// (tile, chunk, lane) writes 16 contiguous bf16 = one lane's A fragment.
// ---------------------------------------------------------------------------
__global__ void repack_gemm_w_kernel(const float* __restrict__ w,
                                     unsigned short* __restrict__ out,
                                     int Cout, int Cin, int nTiles)
{
  int idx = blockIdx.x * 256 + threadIdx.x;
  int nChunk = Cin >> 5;
  int total = nTiles * nChunk * 32;
  if (idx >= total) return;
  int lane = idx & 31;
  int t2 = idx >> 5;
  int ci = t2 % nChunk;
  int tile = t2 / nChunk;
  int half = lane >> 4;
  int m = tile * 16 + (lane & 15);
  unsigned short* o = out + (size_t)idx * 16;
#pragma unroll
  for (int e = 0; e < 16; ++e) {
    int K = a_frag_k(e, half);
    float val = (m < Cout) ? w[(size_t)m * Cin + ci * 32 + K] : 0.0f;
    o[e] = (unsigned short)f32_to_bf16_bits(val);
  }
}

// Conv weight repack: OIHW (192,384,3,3) -> [tile(12)][tap(9)][chunk(12)][lane][16e]
__global__ void repack_conv_w_kernel(const float* __restrict__ w,
                                     unsigned short* __restrict__ out)
{
  int idx = blockIdx.x * 256 + threadIdx.x;     // 12*9*12*32 = 41472
  if (idx >= 12 * 9 * 12 * 32) return;
  int lane = idx & 31;
  int r = idx >> 5;
  int ci = r % 12;  r /= 12;
  int tap = r % 9;
  int tile = r / 9;
  int half = lane >> 4;
  int m = tile * 16 + (lane & 15);              // < 192 always
  unsigned short* o = out + (size_t)idx * 16;
#pragma unroll
  for (int e = 0; e < 16; ++e) {
    int K = a_frag_k(e, half);
    o[e] = (unsigned short)f32_to_bf16_bits(
        w[((size_t)m * (2 * CCH) + ci * 32 + K) * 9 + tap]);
  }
}

// ---------------------------------------------------------------------------
// Per-pixel channel GEMM with packed weights.
// Block: 4 waves x 16 out-channels, 32-pixel tile. Per 32-K chunk:
// stage activations (bf16-pair packed 32-bit LDS stores), each wave loads
// one 32B A fragment from the packed stream and issues 2 WMMAs (2 B halves).
// ---------------------------------------------------------------------------
__global__ __launch_bounds__(128) void channel_gemm_kernel(
    const float* __restrict__ in, const unsigned short* __restrict__ wp,
    const float* __restrict__ bias, float* __restrict__ out,
    int Cout, int Cin, float scale, int relu)
{
  __shared__ __align__(64) unsigned int lds32[32 * 16];   // [pixel][k-pair]
  const int wave = threadIdx.x >> 5;
  const int lane = threadIdx.x & 31;
  const int half = lane >> 4;
  const int pN   = lane & 15;
  const int b    = blockIdx.z;
  const int pixBase = blockIdx.x * 32;
  const int mtile = blockIdx.y * 4 + wave;
  const int nChunk = Cin >> 5;
  const float* inB = in + (size_t)b * Cin * HWSZ;
  const unsigned short* ldsH = (const unsigned short*)lds32;

  v8f acc0 = {}, acc1 = {};
  for (int ci = 0; ci < nChunk; ++ci) {
    const int kb = ci * 32;
    __syncthreads();
    for (int e = threadIdx.x; e < 512; e += 128) {
      int k2 = e >> 5, p = e & 31;
      float f0 = inB[(size_t)(kb + 2 * k2 + 0) * HWSZ + pixBase + p];
      float f1 = inB[(size_t)(kb + 2 * k2 + 1) * HWSZ + pixBase + p];
      lds32[p * 16 + k2] = pack_bf16x2(f0, f1);
    }
    __syncthreads();
    v16bf afrag = *(const v16bf*)(wp + ((size_t)(mtile * nChunk + ci) * 32 + lane) * 16);
    v16bf b0 = *(const v16bf*)(ldsH + (pN * 32 + half * 16));
    v16bf b1 = *(const v16bf*)(ldsH + ((pN + 16) * 32 + half * 16));
    acc0 = __builtin_amdgcn_wmma_f32_16x16x32_bf16(
        false, afrag, false, b0, (short)0, acc0, false, false);
    acc1 = __builtin_amdgcn_wmma_f32_16x16x32_bf16(
        false, afrag, false, b1, (short)0, acc1, false, false);
  }
#pragma unroll
  for (int r = 0; r < 8; ++r) {
    int M = mtile * 16 + r + half * 8;        // C/D layout: lanes 16-31 -> M+8
    if (M < Cout) {
      float bv = bias ? bias[M] : 0.0f;
      float o0 = acc0[r] * scale + bv;
      float o1 = acc1[r] * scale + bv;
      if (relu) { o0 = fmaxf(o0, 0.0f); o1 = fmaxf(o1, 0.0f); }
      float* op = out + ((size_t)b * Cout + M) * HWSZ + pixBase + pN;
      op[0]  = o0;
      op[16] = o1;
    }
  }
}

// ---------------------------------------------------------------------------
// 3x3 SAME conv (384->192) + bias + relu, implicit WMMA GEMM.
// Block: one row y, 32 x's, 4 waves x 16 outch. Per (row-tap i, 32-ch chunk):
// stage a 34-pixel halo slab once, reuse for 3 column taps x 2 pixel halves.
// ---------------------------------------------------------------------------
__global__ __launch_bounds__(128) void conv3x3_relu_kernel(
    const float* __restrict__ xq, const float* __restrict__ xkv,
    const unsigned short* __restrict__ wp, const float* __restrict__ bias,
    float* __restrict__ out)
{
  __shared__ __align__(64) unsigned int lds32[34 * 16];   // [pixel][k-pair]
  const int wave = threadIdx.x >> 5;
  const int lane = threadIdx.x & 31;
  const int half = lane >> 4;
  const int pN   = lane & 15;
  const int b    = blockIdx.z;
  const int y    = blockIdx.x >> 2;
  const int x0   = (blockIdx.x & 3) * 32;
  const int mtile = blockIdx.y * 4 + wave;
  const unsigned short* ldsH = (const unsigned short*)lds32;

  v8f acc0 = {}, acc1 = {};
  for (int i = 0; i < 3; ++i) {
    const int yy = y + i - 1;
    const bool rowOK = (yy >= 0) && (yy < HH);
    for (int ci = 0; ci < 12; ++ci) {
      const int cb = ci * 32;
      __syncthreads();
      for (int e = threadIdx.x; e < 544; e += 128) {      // 34 px * 16 pairs
        int k2 = e >> 5, p = e % 32 + (e >= 512 ? 0 : 0); // decompose below
        // e = p * 16 + k2 layout instead for full coverage:
        int px = e / 16;
        k2 = e & 15;
        int xx = x0 + px - 1;
        float f0 = 0.0f, f1 = 0.0f;
        if (rowOK && xx >= 0 && xx < WW) {
          int c = cb + 2 * k2;
          const float* src = (c < CCH) ? xq : xkv;
          int cc = (c < CCH) ? c : c - CCH;
          const float* base = src + ((size_t)b * CCH + cc) * HWSZ + yy * WW + xx;
          f0 = base[0];
          f1 = base[HWSZ];          // next channel (pairs never straddle 192)
        }
        lds32[px * 16 + k2] = pack_bf16x2(f0, f1);
        (void)p;
      }
      __syncthreads();
#pragma unroll
      for (int j = 0; j < 3; ++j) {
        v16bf afrag = *(const v16bf*)(
            wp + (((size_t)(mtile * 9 + i * 3 + j) * 12 + ci) * 32 + lane) * 16);
        v16bf b0 = *(const v16bf*)(ldsH + ((pN + j) * 32 + half * 16));
        v16bf b1 = *(const v16bf*)(ldsH + ((pN + 16 + j) * 32 + half * 16));
        acc0 = __builtin_amdgcn_wmma_f32_16x16x32_bf16(
            false, afrag, false, b0, (short)0, acc0, false, false);
        acc1 = __builtin_amdgcn_wmma_f32_16x16x32_bf16(
            false, afrag, false, b1, (short)0, acc1, false, false);
      }
    }
  }
#pragma unroll
  for (int r = 0; r < 8; ++r) {
    int M = mtile * 16 + r + half * 8;
    float bv = bias[M];
    float* op = out + ((size_t)b * CCH + M) * HWSZ + (size_t)y * WW + x0 + pN;
    op[0]  = fmaxf(acc0[r] + bv, 0.0f);
    op[16] = fmaxf(acc1[r] + bv, 0.0f);
  }
}

// ---------------------------------------------------------------------------
// Deformable attention core: one wave per (b, head, pixel); lane = head-local
// channel, c = lane*NHEAD + head. Gather-bound; q.k dot via shfl_xor tree.
// OOB window positions contribute logit 0 / value 0 (matches zero-padding).
// ---------------------------------------------------------------------------
__device__ __forceinline__ float wave_sum32(float x) {
  x += __shfl_xor(x, 16, 32);
  x += __shfl_xor(x,  8, 32);
  x += __shfl_xor(x,  4, 32);
  x += __shfl_xor(x,  2, 32);
  x += __shfl_xor(x,  1, 32);
  return x;
}

__global__ __launch_bounds__(256) void deform_attn_kernel(
    const float* __restrict__ q, const float* __restrict__ k,
    const float* __restrict__ v, const float* __restrict__ off,
    float* __restrict__ outAttn)
{
  const int wave = threadIdx.x >> 5;
  const int lane = threadIdx.x & 31;
  int gw   = blockIdx.x * 8 + wave;
  int pix  = gw % HWSZ;
  int t    = gw / HWSZ;
  int head = t % NHEAD;
  int b    = t / NHEAD;
  int y = pix >> 7, x = pix & (WW - 1);
  int c = lane * NHEAD + head;

  const float* kc   = k + ((size_t)b * CCH + c) * HWSZ;
  const float* vc   = v + ((size_t)b * CCH + c) * HWSZ;
  const float* offb = off + (size_t)b * (2 * NOFF) * HWSZ;
  float qv = q[((size_t)b * CCH + c) * HWSZ + pix];

  float logits[NOFF], vs[NOFF];
#pragma unroll
  for (int idx = 0; idx < NOFF; ++idx) {
    int i = idx / 3, j = idx % 3;
    int yy = y + i - 1, xx = x + j - 1;
    float lg = 0.0f, sv = 0.0f;
    if (yy >= 0 && yy < HH && xx >= 0 && xx < WW) {
      int np = yy * WW + xx;
      float ox = offb[(size_t)(2 * idx + 0) * HWSZ + np];
      float oy = offb[(size_t)(2 * idx + 1) * HWSZ + np];
      float x0f = floorf(ox), y0f = floorf(oy);
      float wx1 = ox - x0f,  wy1 = oy - y0f;
      float kacc = 0.0f, vacc = 0.0f;
#pragma unroll
      for (int cy = 0; cy < 2; ++cy) {
#pragma unroll
        for (int cx = 0; cx < 2; ++cx) {
          float xf = x0f + (float)cx, yf = y0f + (float)cy;
          float wgt = (cx ? wx1 : 1.0f - wx1) * (cy ? wy1 : 1.0f - wy1);
          bool valid = (xf >= 0.0f) && (xf <= (float)(WW - 1)) &&
                       (yf >= 0.0f) && (yf <= (float)(HH - 1));
          int xc = (int)fminf(fmaxf(xf, 0.0f), (float)(WW - 1));
          int yc = (int)fminf(fmaxf(yf, 0.0f), (float)(HH - 1));
          int sp = yc * WW + xc;
          float m = valid ? wgt : 0.0f;
          kacc += m * kc[sp];
          vacc += m * vc[sp];
        }
      }
      lg = wave_sum32(qv * kacc);          // q pre-scaled by Ch^-0.5
      sv = vacc;
    }
    logits[idx] = lg;
    vs[idx] = sv;
  }
  float mx = logits[0];
#pragma unroll
  for (int idx = 1; idx < NOFF; ++idx) mx = fmaxf(mx, logits[idx]);
  float ssum = 0.0f, ex[NOFF];
#pragma unroll
  for (int idx = 0; idx < NOFF; ++idx) { ex[idx] = __expf(logits[idx] - mx); ssum += ex[idx]; }
  float inv = 1.0f / ssum;
  float o = 0.0f;
#pragma unroll
  for (int idx = 0; idx < NOFF; ++idx) o += ex[idx] * inv * vs[idx];
  outAttn[((size_t)b * CCH + c) * HWSZ + pix] = o;
}

// ---------------------------------------------------------------------------
extern "C" void kernel_launch(void* const* d_in, const int* in_sizes, int n_in,
                              void* d_out, int out_size, void* d_ws, size_t ws_size,
                              hipStream_t stream)
{
  (void)in_sizes; (void)n_in; (void)out_size; (void)ws_size;
  const float* x_q    = (const float*)d_in[0];
  const float* x_kv   = (const float*)d_in[1];
  const float* w_q    = (const float*)d_in[2];
  const float* w_kv   = (const float*)d_in[3];
  const float* w_off1 = (const float*)d_in[4];
  const float* b_off1 = (const float*)d_in[5];
  const float* w_off2 = (const float*)d_in[6];
  const float* b_off2 = (const float*)d_in[7];
  const float* w_proj = (const float*)d_in[8];
  const float* b_proj = (const float*)d_in[9];
  float* out = (float*)d_out;
  float* ws  = (float*)d_ws;

  const size_t SZ = (size_t)BATCH * CCH * HWSZ;       // 6,291,456 floats
  float* qb   = ws;
  float* kb   = ws + 1 * SZ;
  float* vb   = ws + 2 * SZ;
  float* h1   = ws + 3 * SZ;
  float* attn = ws + 4 * SZ;
  float* offb = ws + 5 * SZ;                          // + B*18*HWSZ floats
  const size_t FLOATS = 5 * SZ + (size_t)BATCH * 2 * NOFF * HWSZ;

  unsigned short* pk = (unsigned short*)(ws + FLOATS);
  const size_t GW = (size_t)12 * 6 * 32 * 16;         // 12-tile GEMM pack
  const size_t OW = (size_t)4  * 6 * 32 * 16;         // off2 pack (4 tiles)
  unsigned short* wq_p    = pk;
  unsigned short* wk_p    = wq_p + GW;
  unsigned short* wv_p    = wk_p + GW;
  unsigned short* wproj_p = wv_p + GW;
  unsigned short* woff2_p = wproj_p + GW;
  unsigned short* wconv_p = woff2_p + OW;             // 12*9*12*32*16

  // ---- weight repacks (tiny; L2-resident afterwards) ----
  {
    int tg = 12 * 6 * 32;                             // 2304 threads
    repack_gemm_w_kernel<<<dim3((tg + 255) / 256), dim3(256), 0, stream>>>(
        w_q, wq_p, CCH, CCH, 12);
    repack_gemm_w_kernel<<<dim3((tg + 255) / 256), dim3(256), 0, stream>>>(
        w_kv, wk_p, CCH, CCH, 12);
    repack_gemm_w_kernel<<<dim3((tg + 255) / 256), dim3(256), 0, stream>>>(
        w_kv + (size_t)CCH * CCH, wv_p, CCH, CCH, 12);
    repack_gemm_w_kernel<<<dim3((tg + 255) / 256), dim3(256), 0, stream>>>(
        w_proj, wproj_p, CCH, CCH, 12);
    int to = 4 * 6 * 32;                              // 768 threads
    repack_gemm_w_kernel<<<dim3((to + 255) / 256), dim3(256), 0, stream>>>(
        w_off2, woff2_p, 2 * NOFF, CCH, 4);
    int tc = 12 * 9 * 12 * 32;                        // 41472 threads
    repack_conv_w_kernel<<<dim3((tc + 255) / 256), dim3(256), 0, stream>>>(
        w_off1, wconv_p);
  }

  const dim3 blk(128);
  const dim3 gridGemm(HWSZ / 32, 3, BATCH);           // 512 x 3 x 2
  const float scale = 0.17677669529663687f;           // Ch^-0.5, Ch = 32

  channel_gemm_kernel<<<gridGemm, blk, 0, stream>>>(x_q,  wq_p, nullptr, qb, CCH, CCH, scale, 0);
  channel_gemm_kernel<<<gridGemm, blk, 0, stream>>>(x_kv, wk_p, nullptr, kb, CCH, CCH, 1.0f, 0);
  channel_gemm_kernel<<<gridGemm, blk, 0, stream>>>(x_kv, wv_p, nullptr, vb, CCH, CCH, 1.0f, 0);
  conv3x3_relu_kernel<<<dim3(HWSZ / 32, 3, BATCH), blk, 0, stream>>>(x_q, x_kv, wconv_p, b_off1, h1);
  channel_gemm_kernel<<<dim3(HWSZ / 32, 1, BATCH), blk, 0, stream>>>(h1, woff2_p, b_off2, offb, 2 * NOFF, CCH, 1.0f, 0);
  deform_attn_kernel<<<dim3((BATCH * NHEAD * HWSZ) / 8), dim3(256), 0, stream>>>(qb, kb, vb, offb, attn);
  channel_gemm_kernel<<<gridGemm, blk, 0, stream>>>(attn, wproj_p, b_proj, out, CCH, CCH, 1.0f, 0);
}
